// GumbelSoftmaxModule_3272765079610
// MI455X (gfx1250) — matrix operand order repaired and checked by reading, not compiled, not run on hardware
//
#include <hip/hip_runtime.h>
#include <hip/hip_bf16.h>

// ---------------------------------------------------------------------------
// Problem constants (from reference setup_inputs)
// ---------------------------------------------------------------------------
#define BB    16          // batch
#define DD    256         // channels
#define TT    4096        // time
#define KK    512         // codes per codebook
#define SUBD  128         // sub-vector width per codebook
#define NROWS (BB * TT)   // 65536 flattened rows
#define TAU_INV 0.5f
#define EPS_P  1e-10f

typedef __bf16 bf16;
typedef bf16  v16bf __attribute__((ext_vector_type(16)));
typedef bf16  v8bf  __attribute__((ext_vector_type(8)));
typedef float v8f   __attribute__((ext_vector_type(8)));

// ---------------------------------------------------------------------------
// Operand loaders matching CDNA5 16-bit WMMA VGPR layouts (wave32).
// A (16xK, M=lane%16): two contiguous 8-elem K-groups at +(lane/16)*8 and +16.
// B (Kx16, N=lane%16): two contiguous 8-elem K-groups at +(lane/16)*8 and +16.
// ---------------------------------------------------------------------------
__device__ __forceinline__ v16bf load_a16(const bf16* rowp, int kk, int lane) {
  int g = (lane >> 4) * 8;
  v8bf lo = *(const v8bf*)(rowp + kk * 32 + g);
  v8bf hi = *(const v8bf*)(rowp + kk * 32 + g + 16);
  v16bf r;
#pragma unroll
  for (int i = 0; i < 8; ++i) { r[i] = lo[i]; r[i + 8] = hi[i]; }
  return r;
}

__device__ __forceinline__ v16bf load_b16(const bf16* colp, int kk, int lane) {
  int g = (lane >> 4) * 8;
  v8bf lo = *(const v8bf*)(colp + kk * 32 + g);
  v8bf hi = *(const v8bf*)(colp + kk * 32 + g + 16);
  v16bf r;
#pragma unroll
  for (int i = 0; i < 8; ++i) { r[i] = lo[i]; r[i + 8] = hi[i]; }
  return r;
}

// Cheap 2-round counter hash -> (0,1) for Gumbel noise (~5 VALU/elem).
__device__ __forceinline__ float hash_u01(unsigned x) {
  x *= 0x9e3779b9u; x ^= x >> 16;
  x *= 0x85ebca6bu; x ^= x >> 13;
  return ((x >> 8) + 0.5f) * (1.0f / 16777216.0f);
}

// ---------------------------------------------------------------------------
// Prep: bf16 codebooks (row-major + transposed), code norms, zero accumulators.
// ---------------------------------------------------------------------------
__global__ void vq_prep_kernel(const float* __restrict__ emb1,
                               const float* __restrict__ emb2,
                               bf16* __restrict__ embb,
                               bf16* __restrict__ embtb,
                               float* __restrict__ ce,
                               float* __restrict__ avg) {
  int tid    = blockIdx.x * blockDim.x + threadIdx.x;
  int stride = gridDim.x * blockDim.x;
  for (int e = tid; e < 2 * KK * SUBD; e += stride) {
    int c = e / (KK * SUBD);
    int r = e % (KK * SUBD);
    int k = r / SUBD, d = r % SUBD;
    const float* src = c ? emb2 : emb1;
    float v = src[k * SUBD + d];
    embb[e] = (bf16)v;
    embtb[c * KK * SUBD + d * KK + k] = (bf16)v;
  }
  for (int e = tid; e < 2 * KK; e += stride) {
    int c = e / KK, k = e % KK;
    const float* src = c ? emb2 : emb1;
    float s = 0.f;
#pragma unroll 8
    for (int d = 0; d < SUBD; ++d) { float v = src[k * SUBD + d]; s += v * v; }
    ce[e]  = s;
    avg[e] = 0.f;
  }
}

// ---------------------------------------------------------------------------
// Fused main kernel. Block = 256 threads (8 wave32). One block handles a
// 16-row tile for one codebook: dist-GEMM -> gumbel-softmax -> quant-GEMM,
// all intermediates in LDS (no N*K global tensors ever materialized).
// ---------------------------------------------------------------------------
__global__ __launch_bounds__(256) void vq_main_kernel(
    const float* __restrict__ x,
    const bf16*  __restrict__ embb,
    const bf16*  __restrict__ embtb,
    const float* __restrict__ ce,
    float*       __restrict__ avg,
    float*       __restrict__ out) {
  __shared__ bf16  xt[16 * SUBD];       // x tile, bf16, row-major (4 KB)
  __shared__ float logits[16 * KK];     // logits then enc f32 (32 KB)
  __shared__ bf16  encs[16 * KK];       // enc bf16 for GEMM2 (16 KB)
  __shared__ float ces[KK];             // ||e||^2 staged per block (2 KB)

  const int tile = blockIdx.x;          // 0..4095 (16-row tiles)
  const int c    = blockIdx.y;          // codebook 0/1
  const int tid  = threadIdx.x;
  const int lane = tid & 31;
  const int w    = tid >> 5;            // wave 0..7

  const int nrow0 = tile * 16;
  const int b     = nrow0 / TT;
  const int t0    = nrow0 % TT;

  const bf16*  emb_c  = embb  + (size_t)c * KK * SUBD;
  const bf16*  embt_c = embtb + (size_t)c * SUBD * KK;
  const float* ce_c   = ce + c * KK;

  // ---- Stage ||e||^2 table and x tile (16 rows x 128 dims, bf16) into LDS ----
  ces[tid]       = ce_c[tid];
  ces[tid + 256] = ce_c[tid + 256];
  const float* xb = x + ((size_t)b * DD + c * SUBD) * TT + t0;
#pragma unroll
  for (int it = 0; it < 8; ++it) {
    int e = tid + it * 256;
    int r = e & 15, d = e >> 4;
    xt[r * SUBD + d] = (bf16)xb[(size_t)d * TT + r];
  }
  __syncthreads();

  // ---- GEMM1: logits^T tiles = emb @ xtile^T ; wave w owns codes [w*64, w*64+64) ----
  const int cb0 = w * 64;
  for (int mt = 0; mt < 4; ++mt) {
    const int   codeA = cb0 + mt * 16 + (lane & 15);
    const bf16* arow  = emb_c + (size_t)codeA * SUBD;
    const bf16* brow  = xt + (lane & 15) * SUBD;
    __builtin_prefetch(arow, 0, 1);
    v8f acc = {};
#pragma unroll
    for (int kk = 0; kk < 4; ++kk) {
      v16bf a  = load_a16(arow, kk, lane);
      v16bf bm = load_b16(brow, kk, lane);
      acc = __builtin_amdgcn_wmma_f32_16x16x32_bf16(
          false, a, false, bm, (short)0, acc, false, false);
    }
    // Epilogue: logits = (2*dot - ||e||^2 + gumbel) / tau  (||x||^2 cancels in softmax)
    const int row  = lane & 15;
    const int mofs = 8 * (lane >> 4);
#pragma unroll
    for (int i = 0; i < 8; ++i) {
      int   code = cb0 + mt * 16 + mofs + i;
      float dot  = acc[i];
      float u    = hash_u01((unsigned)((nrow0 + row) * 1024 + code * 2 + c));
      float g    = -__logf(-__logf(u));
      logits[row * KK + code] = (2.0f * dot - ces[code] + g) * TAU_INV;
    }
  }
  __syncthreads();

  // ---- Softmax over K=512 per row; wave w handles rows 2w, 2w+1 ----
#pragma unroll
  for (int rr = 0; rr < 2; ++rr) {
    const int row = w * 2 + rr;
    float l[16];
    float m = -1e30f;
#pragma unroll
    for (int j = 0; j < 16; ++j) { l[j] = logits[row * KK + lane + j * 32]; m = fmaxf(m, l[j]); }
    for (int off = 16; off >= 1; off >>= 1) m = fmaxf(m, __shfl_xor(m, off, 32));
    float s = 0.f;
#pragma unroll
    for (int j = 0; j < 16; ++j) { l[j] = __expf(l[j] - m); s += l[j]; }
    for (int off = 16; off >= 1; off >>= 1) s += __shfl_xor(s, off, 32);
    const float inv = 1.0f / s;
#pragma unroll
    for (int j = 0; j < 16; ++j) {
      float e = l[j] * inv;
      logits[row * KK + lane + j * 32] = e;        // f32 enc (for perplexity sum)
      encs[row * KK + lane + j * 32]   = (bf16)e;  // bf16 enc (GEMM2 A operand)
    }
  }
  __syncthreads();

  // ---- Perplexity partial: per-block column sums -> one atomic per code ----
#pragma unroll
  for (int j = 0; j < 2; ++j) {
    int code = tid + j * 256;
    float s = 0.f;
#pragma unroll
    for (int r = 0; r < 16; ++r) s += logits[r * KK + code];
    atomicAdd(&avg[c * KK + code], s);
  }

  // ---- GEMM2: quant(16 x 128) = enc @ emb ; wave w owns d-cols [w*16, w*16+16) ----
  {
    const bf16* arow = encs + (lane & 15) * KK;
    const int   dcol = w * 16 + (lane & 15);
    const bf16* bcol = embt_c + (size_t)dcol * KK;
    __builtin_prefetch(bcol, 0, 1);
    v8f acc = {};
#pragma unroll
    for (int kk = 0; kk < 16; ++kk) {
      v16bf a  = load_a16(arow, kk, lane);
      v16bf bm = load_b16(bcol, kk, lane);
      acc = __builtin_amdgcn_wmma_f32_16x16x32_bf16(
          false, a, false, bm, (short)0, acc, false, false);
    }
    // Output already in (B, D, T) layout: lane -> d column, vgpr i -> t offset.
    // Two explicit 16B stores per lane (addresses 16-row aligned).
    float* ob = out + ((size_t)b * DD + c * SUBD + dcol) * TT + t0 + 8 * (lane >> 4);
    float4 o0 = make_float4(acc[0], acc[1], acc[2], acc[3]);
    float4 o1 = make_float4(acc[4], acc[5], acc[6], acc[7]);
    *(float4*)(ob)     = o0;
    *(float4*)(ob + 4) = o1;
  }
}

// ---------------------------------------------------------------------------
// Finalize perplexity: perp = sum_c exp(-sum_k p log(p+eps)), p = avg/N
// ---------------------------------------------------------------------------
__global__ void vq_perp_kernel(const float* __restrict__ avg, float* __restrict__ outp) {
  __shared__ float red[256];
  const int tid = threadIdx.x;
  float total = 0.f;
  for (int c = 0; c < 2; ++c) {
    float h = 0.f;
    for (int code = tid; code < KK; code += 256) {
      float p = avg[c * KK + code] * (1.0f / (float)NROWS);
      h += p * __logf(p + EPS_P);
    }
    red[tid] = h;
    __syncthreads();
    for (int s = 128; s > 0; s >>= 1) {
      if (tid < s) red[tid] += red[tid + s];
      __syncthreads();
    }
    if (tid == 0) total += __expf(-red[0]);
    __syncthreads();
  }
  if (tid == 0) outp[0] = total;
}

// ---------------------------------------------------------------------------
// Launch
// ---------------------------------------------------------------------------
extern "C" void kernel_launch(void* const* d_in, const int* in_sizes, int n_in,
                              void* d_out, int out_size, void* d_ws, size_t ws_size,
                              hipStream_t stream) {
  (void)in_sizes; (void)n_in; (void)out_size; (void)ws_size;
  const float* x    = (const float*)d_in[0];
  const float* emb1 = (const float*)d_in[1];
  const float* emb2 = (const float*)d_in[2];
  float*       out  = (float*)d_out;

  char* ws = (char*)d_ws;
  bf16*  embb  = (bf16*)ws;                           // 256 KB: bf16 codebooks row-major
  bf16*  embtb = (bf16*)(ws + 256 * 1024);            // 256 KB: transposed (code-contig)
  float* ce    = (float*)(ws + 512 * 1024);           // 4 KB: ||e||^2 per code
  float* avg   = (float*)(ws + 512 * 1024 + 4096);    // 4 KB: enc column sums

  vq_prep_kernel<<<512, 256, 0, stream>>>(emb1, emb2, embb, embtb, ce, avg);

  dim3 grid(NROWS / 16, 2);   // 4096 row tiles x 2 codebooks
  vq_main_kernel<<<grid, 256, 0, stream>>>(x, embb, embtb, ce, avg, out);

  vq_perp_kernel<<<1, 256, 0, stream>>>(avg, out + (size_t)BB * DD * TT);
}
